// RNNModel_42468636623573
// MI455X (gfx1250) — matrix operand order repaired and checked by reading, not compiled, not run on hardware
//
#include <hip/hip_runtime.h>
#include <hip/hip_bf16.h>

// ---------------------------------------------------------------------------
// AWD-LSTM LM forward on gfx1250 (MI455X).
// - All GEMMs: v_wmma_f32_16x16x32_f16, 2x2 register blocking per wave.
// - Recurrent scan: ONE persistent kernel per layer (software grid barrier),
//   h_prev staged to LDS via global_load_async_to_lds_b128 (+ s_wait_asynccnt),
//   A-fragments served from LDS (ds_load_b128), U weights L2-resident.
// ---------------------------------------------------------------------------

typedef __attribute__((ext_vector_type(16))) _Float16 v16h;
typedef __attribute__((ext_vector_type(8)))  _Float16 v8h;
typedef __attribute__((ext_vector_type(8)))  float    v8f;

#define VOCABP1 33279     // emb has VOCAB+1 rows; decode/softmax width
#define NINP    400
#define NHID    1150
#define BATCH   16
#define TLEN    256
#define MROWS   (BATCH * TLEN)   // 4096 token rows, time-major: m = t*16 + b
#define KHMAX   1152             // max padded hidden K

// ---------------------------------------------------------------------------
// WMMA fragment loaders (CDNA5 ISA 7.12.2, wave32)
// ---------------------------------------------------------------------------
__device__ __forceinline__ v16h ldA(const _Float16* __restrict__ A, int lda,
                                    int m0, int k0) {
  int lane = threadIdx.x & 31;
  int row = lane & 15, hi = lane >> 4;
  const _Float16* p = A + (size_t)(m0 + row) * lda + k0 + hi * 8;
  v16h a;
  ((v8h*)&a)[0] = *(const v8h*)p;          // K = hi*8 .. +7
  ((v8h*)&a)[1] = *(const v8h*)(p + 16);   // K = 16+hi*8 .. +7
  return a;
}

__device__ __forceinline__ v16h ldB(const _Float16* __restrict__ B, int ldb,
                                    int n0, int k0) {
  int lane = threadIdx.x & 31;
  int col = lane & 15, hi = lane >> 4;
  const _Float16* p = B + (size_t)(n0 + col) * ldb + k0 + hi * 8;
  v16h b;
  ((v8h*)&b)[0] = *(const v8h*)p;          // K = hi*8 .. +7
  ((v8h*)&b)[1] = *(const v8h*)(p + 16);   // K = 16+hi*8 .. +7
  return b;
}

#define WMMA_F16(a, b, c) \
  __builtin_amdgcn_wmma_f32_16x16x32_f16(false, (a), false, (b), (short)0, (c), false, false)

// ---------------------------------------------------------------------------
// Tiled WMMA GEMM with 2x2 register blocking.
// block = 256 threads (8 waves as 2(M) x 4(N)); block tile = 64M x 128N.
// C[M,N] = A[M,Kpad] * Bp[N,Kpad]^T (+ bias[col]);  store guarded col<Ncols.
// remapBT: row (t*16+b) -> (b*TLEN + t) for the [B,T,V] output.
// ---------------------------------------------------------------------------
__global__ void wmma_gemm(const _Float16* __restrict__ A, int lda,
                          const _Float16* __restrict__ Bp, int ldb,
                          float* __restrict__ C, long ldc,
                          const float* __restrict__ bias,
                          int Ktiles, int Ncols, int remapBT) {
  int wave = threadIdx.x >> 5;
  int lane = threadIdx.x & 31;
  int wm = wave >> 2, wn = wave & 3;
  int m0w = blockIdx.y * 64 + wm * 32;
  int n0w = blockIdx.x * 128 + wn * 32;

  v8f c00 = {}, c01 = {}, c10 = {}, c11 = {};
  for (int kt = 0; kt < Ktiles; ++kt) {
    int k = kt * 32;
    v16h a0 = ldA(A, lda, m0w, k);
    v16h a1 = ldA(A, lda, m0w + 16, k);
    v16h b0 = ldB(Bp, ldb, n0w, k);
    v16h b1 = ldB(Bp, ldb, n0w + 16, k);
    c00 = WMMA_F16(a0, b0, c00);
    c01 = WMMA_F16(a0, b1, c01);
    c10 = WMMA_F16(a1, b0, c10);
    c11 = WMMA_F16(a1, b1, c11);
  }

  int hi = lane >> 4, cl = lane & 15;
  v8f acc[2][2] = {{c00, c01}, {c10, c11}};
#pragma unroll
  for (int i = 0; i < 2; ++i) {
#pragma unroll
    for (int j = 0; j < 2; ++j) {
      int col = n0w + j * 16 + cl;
      if (col >= Ncols) continue;
      float bv = bias ? bias[col] : 0.0f;
#pragma unroll
      for (int r = 0; r < 8; ++r) {
        int m = m0w + i * 16 + r + hi * 8;            // time-major row t*16+b
        long row = remapBT ? (long)(m & (BATCH - 1)) * TLEN + (m >> 4) : (long)m;
        C[row * ldc + col] = acc[i][j][r] + bv;
      }
    }
  }
}

// ---------------------------------------------------------------------------
// Persistent LSTM scan: one block per unit-tile (16 units = 64 gate cols),
// 128 threads (4 waves, one per gate). Loops over all T timesteps with a
// software grid barrier.  h_prev staged to LDS via async load each step.
// ---------------------------------------------------------------------------
__device__ __forceinline__ void grid_barrier(unsigned* cnt, unsigned* gen,
                                             int t, unsigned nb) {
  __syncthreads();
  if (threadIdx.x == 0) {
    __threadfence();
    unsigned arr = atomicAdd(cnt, 1u) + 1u;
    if (arr == nb * (unsigned)(t + 1)) {
      __threadfence();
      atomicExch(gen, (unsigned)(t + 1));
    } else {
      while (*(volatile unsigned*)gen < (unsigned)(t + 1))
        __builtin_amdgcn_s_sleep(1);
    }
    __threadfence();
  }
  __syncthreads();
}

__global__ void lstm_scan(const float* __restrict__ xg, int ldxg,
                          _Float16* __restrict__ h0,   // ping (t even: prev)
                          _Float16* __restrict__ h1,   // pong
                          const _Float16* __restrict__ Up, int ldu,
                          float* __restrict__ cst,
                          _Float16* __restrict__ hs,
                          int U, int Ktiles, int ldh,
                          unsigned* __restrict__ barcnt,
                          unsigned* __restrict__ bargen) {
  __shared__ __align__(16) _Float16 hsh[16 * KHMAX];   // 36 KB
  __shared__ float gbuf[4][16][16];                    // 4 KB

  int wave = threadIdx.x >> 5;
  int lane = threadIdx.x & 31;
  int ut = blockIdx.x;
  int n0 = ut * 64 + wave * 16;
  int hi = lane >> 4, col = lane & 15;
  unsigned nb = gridDim.x;
  uint32_t ldsbase = (uint32_t)(size_t)&hsh[0];        // LDS byte offset
  int hbytes = 16 * ldh * (int)sizeof(_Float16);

  for (int t = 0; t < TLEN; ++t) {
    const _Float16* hp = (t & 1) ? h1 : h0;
    _Float16* hc = (t & 1) ? h0 : h1;

    // ---- async-stage h_prev (contiguous 16 x ldh f16) into LDS ----
    {
      uint64_t src = (uint64_t)(size_t)hp;
      for (int o = (int)threadIdx.x * 16; o < hbytes; o += 128 * 16) {
        uint32_t dst = ldsbase + (uint32_t)o;
        uint64_t g = src + (uint64_t)o;
        asm volatile("global_load_async_to_lds_b128 %0, %1, off"
                     :: "v"(dst), "v"(g) : "memory");
      }
      asm volatile("s_wait_asynccnt 0x0" ::: "memory");
      __syncthreads();
    }

    // ---- gate tile = xg[t] + h_prev @ Up (A from LDS, B from L2) ----
    v8f c;
#pragma unroll
    for (int r = 0; r < 8; ++r) {
      int row = t * 16 + r + hi * 8;
      c[r] = xg[(size_t)row * ldxg + n0 + col];
    }
    for (int kt = 0; kt < Ktiles; ++kt) {
      int k = kt * 32;
      if (lane == 0 && kt + 1 < Ktiles)
        __builtin_prefetch(&Up[(size_t)n0 * ldu + k + 32], 0, 0);
      int abase = (lane & 15) * ldh + k + hi * 8;
      v16h a;
      ((v8h*)&a)[0] = *(const v8h*)&hsh[abase];        // ds_load_b128
      ((v8h*)&a)[1] = *(const v8h*)&hsh[abase + 16];
      v16h b = ldB(Up, ldu, n0, k);
      c = WMMA_F16(a, b, c);
    }
#pragma unroll
    for (int r = 0; r < 8; ++r) gbuf[wave][r + hi * 8][col] = c[r];
    __syncthreads();

    // ---- pointwise LSTM update ----
    for (int cell = threadIdx.x; cell < 256; cell += 128) {
      int b = cell >> 4, j = cell & 15;
      int u = ut * 16 + j;
      if (u < U) {
        float gi = gbuf[0][b][j], gf = gbuf[1][b][j];
        float gc = gbuf[2][b][j], go = gbuf[3][b][j];
        float ig = 1.0f / (1.0f + __expf(-gi));
        float fg = 1.0f / (1.0f + __expf(-gf));
        float og = 1.0f / (1.0f + __expf(-go));
        float cg = tanhf(gc);
        float cold = cst[(size_t)b * ldh + u];
        float cn = fg * cold + ig * cg;
        cst[(size_t)b * ldh + u] = cn;
        float h = og * tanhf(cn);
        hc[(size_t)b * ldh + u] = (_Float16)h;
        hs[(size_t)(t * 16 + b) * ldh + u] = (_Float16)h;
      }
    }

    grid_barrier(barcnt, bargen, t, nb);
  }
}

// ---------------------------------------------------------------------------
// Prepack / utility kernels
// ---------------------------------------------------------------------------
__global__ void prepack_W(const float* __restrict__ W, _Float16* __restrict__ Wp,
                          int din, int U, int Npad, int Kpad) {
  int idx = blockIdx.x * blockDim.x + threadIdx.x;
  int total = Npad * Kpad;
  if (idx >= total) return;
  int nc = idx / Kpad, k = idx % Kpad;
  int ut = nc >> 6, gate = (nc >> 4) & 3, j = nc & 15;
  int u = ut * 16 + j;
  float v = (u < U && k < din) ? W[(size_t)k * (4 * U) + gate * U + u] : 0.0f;
  Wp[idx] = (_Float16)v;
}

__global__ void prepack_bias(const float* __restrict__ b, float* __restrict__ bp,
                             int U, int Npad) {
  int nc = blockIdx.x * blockDim.x + threadIdx.x;
  if (nc >= Npad) return;
  int ut = nc >> 6, gate = (nc >> 4) & 3, j = nc & 15;
  int u = ut * 16 + j;
  bp[nc] = (u < U) ? b[gate * U + u] : 0.0f;
}

__global__ void prepack_emb(const float* __restrict__ emb,
                            _Float16* __restrict__ ef, int Npad, int Kpad) {
  int idx = blockIdx.x * blockDim.x + threadIdx.x;
  int total = Npad * Kpad;
  if (idx >= total) return;
  int v = idx / Kpad, k = idx % Kpad;
  ef[idx] = (v < VOCABP1 && k < NINP) ? (_Float16)emb[(size_t)v * NINP + k]
                                      : (_Float16)0.0f;
}

__global__ void embed_gather(const int* __restrict__ tok,
                             const float* __restrict__ emb,
                             _Float16* __restrict__ x0, int Kpad) {
  int idx = blockIdx.x * blockDim.x + threadIdx.x;
  int total = MROWS * Kpad;
  if (idx >= total) return;
  int m = idx / Kpad, d = idx % Kpad;
  int b = m & 15, t = m >> 4;
  int token = tok[b * TLEN + t];
  x0[idx] = (d < NINP) ? (_Float16)emb[(size_t)token * NINP + d]
                       : (_Float16)0.0f;
}

__global__ void fill_h16(_Float16* p, long n) {
  long i = blockIdx.x * (long)blockDim.x + threadIdx.x;
  if (i < n) p[i] = (_Float16)0.0f;
}
__global__ void fill_u32(unsigned* p, long n) {
  long i = blockIdx.x * (long)blockDim.x + threadIdx.x;
  if (i < n) p[i] = 0u;
}
__global__ void fill_f32(float* p, long n) {
  long i = blockIdx.x * (long)blockDim.x + threadIdx.x;
  if (i < n) p[i] = 0.0f;
}

// In-place row softmax: one block (256 thr) per row of [4096, VOCABP1]
__global__ void softmax_rows(float* __restrict__ out, int ncols) {
  __shared__ float red[256];
  float* p = out + (size_t)blockIdx.x * ncols;
  float mx = -3.4e38f;
  for (int i = threadIdx.x; i < ncols; i += 256) mx = fmaxf(mx, p[i]);
  red[threadIdx.x] = mx;
  __syncthreads();
  for (int s = 128; s > 0; s >>= 1) {
    if (threadIdx.x < s) red[threadIdx.x] = fmaxf(red[threadIdx.x], red[threadIdx.x + s]);
    __syncthreads();
  }
  mx = red[0];
  __syncthreads();
  float sum = 0.0f;
  for (int i = threadIdx.x; i < ncols; i += 256) {
    float e = __expf(p[i] - mx);
    p[i] = e;
    sum += e;
  }
  red[threadIdx.x] = sum;
  __syncthreads();
  for (int s = 128; s > 0; s >>= 1) {
    if (threadIdx.x < s) red[threadIdx.x] += red[threadIdx.x + s];
    __syncthreads();
  }
  float inv = 1.0f / red[0];
  for (int i = threadIdx.x; i < ncols; i += 256) p[i] *= inv;
}

// ---------------------------------------------------------------------------
// Host orchestration
// ---------------------------------------------------------------------------
static inline int cdiv_i(int a, int b) { return (a + b - 1) / b; }
static inline int pad_i(int x, int a) { return cdiv_i(x, a) * a; }

extern "C" void kernel_launch(void* const* d_in, const int* in_sizes, int n_in,
                              void* d_out, int out_size, void* d_ws, size_t ws_size,
                              hipStream_t stream) {
  (void)in_sizes; (void)n_in; (void)out_size; (void)ws_size;
  const int*   tokens = (const int*)d_in[0];
  const float* emb    = (const float*)d_in[1];
  const float* Wm[3]  = {(const float*)d_in[2], (const float*)d_in[5], (const float*)d_in[8]};
  const float* Um[3]  = {(const float*)d_in[3], (const float*)d_in[6], (const float*)d_in[9]};
  const float* bm[3]  = {(const float*)d_in[4], (const float*)d_in[7], (const float*)d_in[10]};
  float* out = (float*)d_out;

  const int din[3] = {NINP, NHID, NHID};
  const int Uu[3]  = {NHID, NHID, NINP};
  int Kin[3], Kh[3], Ut[3], Npad[3], Nalloc[3];
  for (int l = 0; l < 3; ++l) {
    Kin[l]    = pad_i(din[l], 32);     // 416, 1152, 1152
    Kh[l]     = pad_i(Uu[l], 32);      // 1152, 1152, 416
    Ut[l]     = cdiv_i(Uu[l], 16);     // 72, 72, 25
    Npad[l]   = Ut[l] * 64;            // 4608, 4608, 1600
    Nalloc[l] = pad_i(Npad[l], 128);   // 4608, 4608, 1664
  }
  const int VN = pad_i(VOCABP1, 128);  // 33280

  // bump allocator in d_ws
  char* base = (char*)d_ws;
  size_t off = 0;
  auto alloc = [&](size_t bytes) -> void* {
    void* p = base + off;
    off = (off + bytes + 255) & ~(size_t)255;
    return p;
  };

  _Float16* embf = (_Float16*)alloc((size_t)VN * 416 * sizeof(_Float16));
  _Float16* x0f  = (_Float16*)alloc((size_t)MROWS * Kin[0] * sizeof(_Float16));
  _Float16 *Wp[3], *Up[3], *hsq[3], *hping[3], *hpong[3];
  float *bp[3], *cst[3];
  for (int l = 0; l < 3; ++l) {
    Wp[l]    = (_Float16*)alloc((size_t)Nalloc[l] * Kin[l] * sizeof(_Float16));
    Up[l]    = (_Float16*)alloc((size_t)Npad[l] * Kh[l] * sizeof(_Float16));
    bp[l]    = (float*)alloc((size_t)Npad[l] * sizeof(float));
    cst[l]   = (float*)alloc((size_t)BATCH * Kh[l] * sizeof(float));
    hping[l] = (_Float16*)alloc((size_t)BATCH * Kh[l] * sizeof(_Float16));
    hpong[l] = (_Float16*)alloc((size_t)BATCH * Kh[l] * sizeof(_Float16));
    hsq[l]   = (_Float16*)alloc((size_t)MROWS * Kh[l] * sizeof(_Float16));
  }
  float*    xg  = (float*)alloc((size_t)MROWS * 4608 * sizeof(float));
  unsigned* bar = (unsigned*)alloc(8 * sizeof(unsigned));  // cnt, gen

  const int TPB = 256;
  auto g1 = [&](long n) { return dim3((unsigned)((n + TPB - 1) / TPB)); };

  // --- prepack ---
  prepack_emb<<<g1((long)VN * 416), TPB, 0, stream>>>(emb, embf, VN, 416);
  embed_gather<<<g1((long)MROWS * Kin[0]), TPB, 0, stream>>>(tokens, emb, x0f, Kin[0]);
  for (int l = 0; l < 3; ++l) {
    prepack_W<<<g1((long)Nalloc[l] * Kin[l]), TPB, 0, stream>>>(Wm[l], Wp[l], din[l], Uu[l], Nalloc[l], Kin[l]);
    prepack_W<<<g1((long)Npad[l] * Kh[l]),  TPB, 0, stream>>>(Um[l], Up[l], Uu[l], Uu[l], Npad[l], Kh[l]);
    prepack_bias<<<g1(Npad[l]), TPB, 0, stream>>>(bm[l], bp[l], Uu[l], Npad[l]);
  }

  // --- layers ---
  for (int l = 0; l < 3; ++l) {
    fill_f32<<<g1((long)BATCH * Kh[l]), TPB, 0, stream>>>(cst[l], (long)BATCH * Kh[l]);
    fill_h16<<<g1((long)BATCH * Kh[l]), TPB, 0, stream>>>(hping[l], (long)BATCH * Kh[l]);
    fill_h16<<<g1((long)BATCH * Kh[l]), TPB, 0, stream>>>(hpong[l], (long)BATCH * Kh[l]);
    fill_h16<<<g1((long)MROWS * Kh[l]), TPB, 0, stream>>>(hsq[l], (long)MROWS * Kh[l]);
    fill_u32<<<1, TPB, 0, stream>>>(bar, 8);

    // xg = X @ Wp^T + b
    const _Float16* Xin = (l == 0) ? x0f : hsq[l - 1];
    dim3 gg(Nalloc[l] / 128, MROWS / 64);
    wmma_gemm<<<gg, 256, 0, stream>>>(Xin, Kin[l], Wp[l], Kin[l], xg,
                                      (long)Npad[l], bp[l], Kin[l] / 32,
                                      Npad[l], 0);

    // persistent sequential scan (single launch for all T steps)
    lstm_scan<<<Ut[l], 128, 0, stream>>>(xg, Npad[l], hping[l], hpong[l],
                                         Up[l], Kh[l], cst[l], hsq[l],
                                         Uu[l], Kh[l] / 32, Kh[l],
                                         bar, bar + 1);
  }

  // --- tied decode into d_out (remap rows to [B,T]) + softmax in place ---
  dim3 gd(VN / 128, MROWS / 64);
  wmma_gemm<<<gd, 256, 0, stream>>>(hsq[2], Kh[2], embf, 416, out,
                                    (long)VOCABP1, nullptr, Kh[2] / 32,
                                    VOCABP1, 1);
  softmax_rows<<<MROWS, 256, 0, stream>>>(out, VOCABP1);
}
